// DeepViT_52819507806285
// MI455X (gfx1250) — matrix operand order repaired and checked by reading, not compiled, not run on hardware
//
#include <hip/hip_runtime.h>

// ---------------- types ----------------
typedef __bf16 bf16;
typedef __attribute__((ext_vector_type(16))) __bf16 v16bf;
typedef __attribute__((ext_vector_type(8)))  __bf16 v8bf;
typedef __attribute__((ext_vector_type(8)))  float  v8f;
typedef __attribute__((ext_vector_type(4)))  unsigned v4u;
typedef __attribute__((ext_vector_type(8)))  int v8i;
typedef __attribute__((ext_vector_type(4)))  int v4i;

#define DEPTH 6
#define HEADS 12
#define DHEAD 64
#define DIM 768
#define MLPD 3072
#define FEAT 1000
#define BB 32
#define NTOK 197          // 196 + cls
#define NPAD 224          // attention j / K padded to multiple of 32
#define ROWS (BB*NTOK)    // 6304
#define ZBH (BB*HEADS)    // 384

static __device__ __forceinline__ bf16 to_bf16(float f) {
  unsigned u = __builtin_bit_cast(unsigned, f);
  unsigned r = (u + 0x7FFFu + ((u >> 16) & 1u)) >> 16;
  return __builtin_bit_cast(bf16, (unsigned short)r);
}
static __device__ __forceinline__ v8bf zero8() {
  bf16 z = __builtin_bit_cast(bf16, (unsigned short)0);
  v8bf v;
#pragma unroll
  for (int i = 0; i < 8; ++i) v[i] = z;
  return v;
}

// ---------------- TDM 2D tile load (bf16, padded LDS rows) ----------------
// Loads tile_rows x tile_cols (bf16 elements) from gsrc (row stride = stride_elems)
// into LDS at lds_addr. rem_rows/rem_cols bound the tensor for hardware OOB
// zero-fill. LDS padding: +4 dwords after every 16 dwords (i.e. 64B row -> 80B
// row = 40-element stride), matching the fragment-read layout.
static __device__ __forceinline__ void tdm_load_tile_bf16(
    unsigned lds_addr, const bf16* gsrc, int tile_cols, int tile_rows,
    unsigned rem_cols, unsigned rem_rows, unsigned stride_elems) {
  unsigned long long ga = (unsigned long long)(const void*)gsrc;
  v4u g0;
  g0[0] = 1u;                                             // count=1 (user D#)
  g0[1] = lds_addr;                                       // lds byte address
  g0[2] = (unsigned)ga;                                   // global_addr[31:0]
  g0[3] = (unsigned)((ga >> 32) & 0x1FFFFFFu) | (2u << 30); // ga[56:32] | type=2
  v8i g1;
  // data_size=1 (2B) [17:16]; pad_enable [20]; pad_interval=3 (16 dw) [24:22];
  // pad_amount=3 (4 dw) [31:25]
  g1[0] = (int)((1u << 16) | (1u << 20) | (3u << 22) | (3u << 25));
  g1[1] = (int)((rem_cols & 0xFFFFu) << 16);              // tensor_dim0[15:0]
  g1[2] = (int)(((rem_cols >> 16) & 0xFFFFu) | ((rem_rows & 0xFFFFu) << 16));
  g1[3] = (int)(((rem_rows >> 16) & 0xFFFFu) | (((unsigned)tile_cols & 0xFFFFu) << 16));
  g1[4] = (int)((unsigned)tile_rows & 0xFFFFu);           // tile_dim1 (tile_dim2=0)
  g1[5] = (int)stride_elems;                              // tensor_dim0_stride[31:0]
  g1[6] = 0;
  g1[7] = 0;
  v4i zz = {0, 0, 0, 0};
#if __has_include(<hip/amd_detail/amd_gfx1250_TDM.h>)
  v8i z8 = {0, 0, 0, 0, 0, 0, 0, 0};
  __builtin_amdgcn_tensor_load_to_lds(g0, g1, zz, zz, z8, 0);
#else
  __builtin_amdgcn_tensor_load_to_lds(g0, g1, zz, zz, 0);
#endif
}

// ---------------- bf16 WMMA GEMM ----------------
// C[M x N] = alpha * A[M x K](bf16) * B(bf16) (+ bias) (+ residual) (+ gelu)
// A row-major (lda), staged per 128x32 tile by the Tensor Data Mover
// (double-buffered, TENSORcnt-synchronized). B: BCOL=false -> row-major K x N;
// BCOL=true -> N x K row-major. Block tile 128x64, 128 threads (4 waves), each
// wave a 32x64 tile = 2x4 WMMA 16x16x32 (8 WMMA / wave / K-step).
// MODE: 0 = f32 store, 1 = +bias f32, 2 = +bias+gelu bf16, 3 = +bias+residual f32
#define BM 128
#define BN 64
#define BK 32
#define LDS_A 40
#define LDS_B 40

template <int MODE, bool BCOL>
__global__ __launch_bounds__(128) void gemm_bf16(
    const bf16* __restrict__ A, int lda, long sA,
    const bf16* __restrict__ B, int ldb, long sB,
    float* __restrict__ Cf, bf16* __restrict__ Cb, int ldc, long sC,
    const float* __restrict__ bias, const float* __restrict__ resid,
    int M, int N, int K, float alpha) {
  __shared__ __attribute__((aligned(16))) bf16 As[2][BM * LDS_A];
  __shared__ __attribute__((aligned(16))) bf16 Bs[2][BN * LDS_B];

  long z = blockIdx.z;
  A += z * sA;
  B += z * sB;
  if (MODE == 2) { Cb += z * sC; } else { Cf += z * sC; }

  const int blockM = blockIdx.y * BM;
  const int blockN = blockIdx.x * BN;
  const int tid = threadIdx.x;
  const int lane = tid & 31;
  const int wave = tid >> 5;
  const int wm = wave * 32;
  const int lrow = lane & 15;
  const bool hiHalf = lane >= 16;

  v8f acc[2][4];
#pragma unroll
  for (int i = 0; i < 2; ++i)
#pragma unroll
    for (int j = 0; j < 4; ++j)
#pragma unroll
      for (int r = 0; r < 8; ++r) acc[i][j][r] = 0.0f;

  // ---- B tile staging: Bs[buf][n][k], N-major so K-fragments are contiguous
  auto stageB = [&](int kt, int buf) {
    const int k0 = kt * BK;
    if (BCOL) {
#pragma unroll
      for (int j = 0; j < 2; ++j) {
        int vec = j * 128 + tid;
        int n = vec >> 2;
        int c8 = (vec & 3) * 8;
        int gn = blockN + n;
        v8bf v = (gn < N) ? *(const v8bf*)(B + (long)gn * ldb + k0 + c8) : zero8();
        *(v8bf*)&Bs[buf][n * LDS_B + c8] = v;
      }
    } else {
#pragma unroll
      for (int j = 0; j < 2; ++j) {
        int vec = j * 128 + tid;
        int kr = vec >> 3;
        int c8 = (vec & 7) * 8;
        int gc = blockN + c8;
        v8bf v = (gc < N) ? *(const v8bf*)(B + (long)(k0 + kr) * ldb + gc) : zero8();
#pragma unroll
        for (int e = 0; e < 8; ++e) Bs[buf][(c8 + e) * LDS_B + kr] = v[e];
      }
    }
  };
  // ---- A tile staging via TDM (wave 0 issues the DMA; OOB rows zero-filled)
  auto stageA = [&](int kt, int buf) {
    if (tid < 32) {
      tdm_load_tile_bf16((unsigned)(size_t)&As[buf][0],
                         A + (long)blockM * lda + kt * BK, BK, BM,
                         (unsigned)(K - kt * BK), (unsigned)(M - blockM),
                         (unsigned)lda);
    }
  };

  const int ksteps = K / BK;
  stageB(0, 0);
  stageA(0, 0);
  if (tid < 32) __builtin_amdgcn_s_wait_tensorcnt(0);
  __syncthreads();

  for (int kt = 0; kt < ksteps; ++kt) {
    const int cur = kt & 1;
    const int nxt = cur ^ 1;
    if (kt + 1 < ksteps) {   // prefetch next tiles while computing this one
      stageB(kt + 1, nxt);
      stageA(kt + 1, nxt);
    }

    v16bf afrag[2], bfrag[4];
#pragma unroll
    for (int i = 0; i < 2; ++i) {
      int m = wm + i * 16 + lrow;
      int kb = hiHalf ? 8 : 0;
      v8bf lo = *(const v8bf*)&As[cur][m * LDS_A + kb];
      v8bf h8 = *(const v8bf*)&As[cur][m * LDS_A + kb + 16];
#pragma unroll
      for (int e = 0; e < 8; ++e) { afrag[i][e] = lo[e]; afrag[i][e + 8] = h8[e]; }
    }
#pragma unroll
    for (int j = 0; j < 4; ++j) {
      int n = j * 16 + lrow;
      int kb = hiHalf ? 8 : 0;
      v8bf lo = *(const v8bf*)&Bs[cur][n * LDS_B + kb];
      v8bf h8 = *(const v8bf*)&Bs[cur][n * LDS_B + kb + 16];
#pragma unroll
      for (int e = 0; e < 8; ++e) { bfrag[j][e] = lo[e]; bfrag[j][e + 8] = h8[e]; }
    }
#pragma unroll
    for (int i = 0; i < 2; ++i)
#pragma unroll
      for (int j = 0; j < 4; ++j)
        acc[i][j] = __builtin_amdgcn_wmma_f32_16x16x32_bf16(
            false, afrag[i], false, bfrag[j], (short)0, acc[i][j], false, false);

    if (tid < 32) __builtin_amdgcn_s_wait_tensorcnt(0);  // drain TDM before barrier
    __syncthreads();
  }

  // ---- epilogue ----
#pragma unroll
  for (int i = 0; i < 2; ++i)
#pragma unroll
    for (int j = 0; j < 4; ++j)
#pragma unroll
      for (int r = 0; r < 8; ++r) {
        int row = blockM + wm + i * 16 + (hiHalf ? r + 8 : r);
        int col = blockN + j * 16 + lrow;
        if (row < M && col < N) {
          float v = acc[i][j][r] * alpha;
          if (MODE >= 1) v += bias[col];
          long idx = (long)row * ldc + col;
          if (MODE == 2) {
            v = 0.5f * v * (1.0f + erff(v * 0.70710678118654752f));
            Cb[idx] = to_bf16(v);
          } else if (MODE == 3) {
            Cf[idx] = v + resid[idx];
          } else {
            Cf[idx] = v;
          }
        }
      }
}

// ---------------- elementwise kernels ----------------
__global__ void k_f32_to_bf16(const float* __restrict__ in, bf16* __restrict__ out, long n) {
  long i = (long)blockIdx.x * 256 + threadIdx.x;
  if (i < n) out[i] = to_bf16(in[i]);
}

__global__ void k_build_x(const float* __restrict__ xin, const float* __restrict__ cls,
                          float* __restrict__ x0) {
  long idx = (long)blockIdx.x * 256 + threadIdx.x;
  if (idx >= (long)ROWS * DIM) return;
  int d = (int)(idx % DIM);
  long t = idx / DIM;
  int n = (int)(t % NTOK);
  int b = (int)(t / NTOK);
  x0[idx] = (n == 0) ? cls[d] : xin[((long)(b * 196 + n - 1)) * DIM + d];
}

// layernorm over DIM=768, writes bf16; rstride = stride between rows in `in`
__global__ void k_layernorm_bf16(const float* __restrict__ in, long rstride,
                                 const float* __restrict__ g, const float* __restrict__ b,
                                 bf16* __restrict__ out, int rows) {
  int row = blockIdx.x;
  if (row >= rows) return;
  const float* p = in + (long)row * rstride;
  __shared__ float red[256];
  int tid = threadIdx.x;
  float s = 0.f, s2 = 0.f;
  for (int d = tid; d < DIM; d += 256) { float v = p[d]; s += v; s2 += v * v; }
  red[tid] = s; __syncthreads();
  for (int o = 128; o > 0; o >>= 1) { if (tid < o) red[tid] += red[tid + o]; __syncthreads(); }
  float mean = red[0] / (float)DIM;
  __syncthreads();
  red[tid] = s2; __syncthreads();
  for (int o = 128; o > 0; o >>= 1) { if (tid < o) red[tid] += red[tid + o]; __syncthreads(); }
  float var = red[0] / (float)DIM - mean * mean;
  float r = rsqrtf(var + 1e-5f);
  for (int d = tid; d < DIM; d += 256)
    out[(long)row * DIM + d] = to_bf16((p[d] - mean) * r * g[d] + b[d]);
}

// qkv [ROWS x 2304] f32 -> q,k bf16 [Z,197,64]; v bf16 [Z,224,64] zero-padded
__global__ void k_split_qkv(const float* __restrict__ qkv, bf16* __restrict__ q,
                            bf16* __restrict__ k, bf16* __restrict__ v) {
  long idx = (long)blockIdx.x * 256 + threadIdx.x;
  if (idx >= (long)ZBH * NPAD * DHEAD) return;
  int d = (int)(idx % DHEAD);
  long t = idx / DHEAD;
  int n = (int)(t % NPAD);
  int z = (int)(t / NPAD);
  int b = z / HEADS, h = z % HEADS;
  if (n < NTOK) {
    long base = ((long)(b * NTOK + n)) * (3 * DIM) + h * DHEAD + d;
    long qi = ((long)z * NTOK + n) * DHEAD + d;
    q[qi] = to_bf16(qkv[base]);
    k[qi] = to_bf16(qkv[base + DIM]);
    v[idx] = to_bf16(qkv[base + 2 * DIM]);
  } else {
    v[idx] = __builtin_bit_cast(bf16, (unsigned short)0);
  }
}

// in-place softmax over j in [0,197), rows = Z*197, ld = 224, pads zeroed
__global__ void k_softmax(float* __restrict__ s) {
  float* p = s + (long)blockIdx.x * NPAD;
  __shared__ float red[256];
  int tid = threadIdx.x;
  float v = (tid < NTOK) ? p[tid] : -3.4e38f;
  red[tid] = v; __syncthreads();
  for (int o = 128; o > 0; o >>= 1) { if (tid < o) red[tid] = fmaxf(red[tid], red[tid + o]); __syncthreads(); }
  float mx = red[0]; __syncthreads();
  float e = (tid < NTOK) ? __expf(v - mx) : 0.f;
  red[tid] = e; __syncthreads();
  for (int o = 128; o > 0; o >>= 1) { if (tid < o) red[tid] += red[tid + o]; __syncthreads(); }
  float inv = 1.f / red[0];
  if (tid < NPAD) p[tid] = e * inv;
}

// re-attention: mix heads with W[12x12], layernorm across heads, scale/shift, out bf16 padded
__global__ void k_reattn(const float* __restrict__ attn, const float* __restrict__ w,
                         const float* __restrict__ rg, const float* __restrict__ rb,
                         bf16* __restrict__ out) {
  long idx = (long)blockIdx.x * 256 + threadIdx.x;
  if (idx >= (long)BB * NTOK * NPAD) return;
  int j = (int)(idx % NPAD);
  long t = idx / NPAD;
  int i = (int)(t % NTOK);
  int b = (int)(t / NTOK);
  if (j >= NTOK) {
    bf16 zv = __builtin_bit_cast(bf16, (unsigned short)0);
#pragma unroll
    for (int g = 0; g < HEADS; ++g)
      out[(((long)(b * HEADS + g) * NTOK) + i) * NPAD + j] = zv;
    return;
  }
  float a[HEADS];
#pragma unroll
  for (int h = 0; h < HEADS; ++h)
    a[h] = attn[(((long)(b * HEADS + h) * NTOK) + i) * NPAD + j];
  float tt[HEADS]; float m = 0.f;
#pragma unroll
  for (int g = 0; g < HEADS; ++g) {
    float s = 0.f;
#pragma unroll
    for (int h = 0; h < HEADS; ++h) s += w[g * HEADS + h] * a[h];
    tt[g] = s; m += s;
  }
  m *= (1.0f / HEADS);
  float var = 0.f;
#pragma unroll
  for (int g = 0; g < HEADS; ++g) { float d = tt[g] - m; var += d * d; }
  var *= (1.0f / HEADS);
  float r = rsqrtf(var + 1e-5f);
#pragma unroll
  for (int g = 0; g < HEADS; ++g)
    out[(((long)(b * HEADS + g) * NTOK) + i) * NPAD + j] =
        to_bf16((tt[g] - m) * r * rg[g] + rb[g]);
}

// [Z,197,64] f32 -> [ROWS x 768] bf16
__global__ void k_merge_heads(const float* __restrict__ ao, bf16* __restrict__ out) {
  long idx = (long)blockIdx.x * 256 + threadIdx.x;
  if (idx >= (long)ROWS * DIM) return;
  int c = (int)(idx % DIM);
  long t = idx / DIM;
  int n = (int)(t % NTOK);
  int b = (int)(t / NTOK);
  int h = c / DHEAD, d = c % DHEAD;
  out[idx] = to_bf16(ao[(((long)(b * HEADS + h) * NTOK) + n) * DHEAD + d]);
}

// ---------------- host ----------------
static inline long cdiv(long a, long b) { return (a + b - 1) / b; }

extern "C" void kernel_launch(void* const* d_in, const int* in_sizes, int n_in,
                              void* d_out, int out_size, void* d_ws, size_t ws_size,
                              hipStream_t stream) {
  const float* x_in   = (const float*)d_in[0];
  const float* cls    = (const float*)d_in[1];
  const float* ln1_g  = (const float*)d_in[2];
  const float* ln1_b  = (const float*)d_in[3];
  const float* w_qkv  = (const float*)d_in[4];
  const float* re_w   = (const float*)d_in[5];
  const float* re_g   = (const float*)d_in[6];
  const float* re_b   = (const float*)d_in[7];
  const float* w_out  = (const float*)d_in[8];
  const float* b_out  = (const float*)d_in[9];
  const float* ln2_g  = (const float*)d_in[10];
  const float* ln2_b  = (const float*)d_in[11];
  const float* w1     = (const float*)d_in[12];
  const float* b1     = (const float*)d_in[13];
  const float* w2     = (const float*)d_in[14];
  const float* b2     = (const float*)d_in[15];
  const float* lnf_g  = (const float*)d_in[16];
  const float* lnf_b  = (const float*)d_in[17];
  const float* w_head = (const float*)d_in[18];
  const float* b_head = (const float*)d_in[19];
  float* outp = (float*)d_out;

  char* ws = (char*)d_ws;
  size_t off = 0;
  auto alloc = [&](size_t bytes) -> void* {
    off = (off + 255) & ~(size_t)255;
    void* p = ws + off;
    off += bytes;
    return p;
  };

  bf16* wqkv_b  = (bf16*)alloc((size_t)DEPTH * DIM * 3 * DIM * 2);
  bf16* wout_b  = (bf16*)alloc((size_t)DEPTH * DIM * DIM * 2);
  bf16* w1_b    = (bf16*)alloc((size_t)DEPTH * DIM * MLPD * 2);
  bf16* w2_b    = (bf16*)alloc((size_t)DEPTH * MLPD * DIM * 2);
  bf16* whead_b = (bf16*)alloc((size_t)DIM * FEAT * 2);
  float* x0     = (float*)alloc((size_t)ROWS * DIM * 4);
  float* x1     = (float*)alloc((size_t)ROWS * DIM * 4);
  bf16* hbuf    = (bf16*)alloc((size_t)ROWS * DIM * 2);
  float* qkvbuf = (float*)alloc((size_t)ROWS * 3 * DIM * 4);
  bf16* qb      = (bf16*)alloc((size_t)ZBH * NTOK * DHEAD * 2);
  bf16* kb      = (bf16*)alloc((size_t)ZBH * NTOK * DHEAD * 2);
  bf16* vb      = (bf16*)alloc((size_t)ZBH * NPAD * DHEAD * 2);
  float* scores = (float*)alloc((size_t)ZBH * NTOK * NPAD * 4);
  bf16* attn_n  = (bf16*)alloc((size_t)ZBH * NTOK * NPAD * 2);
  float* aout   = (float*)alloc((size_t)ZBH * NTOK * DHEAD * 4);
  bf16* aout_b  = (bf16*)alloc((size_t)ROWS * DIM * 2);
  bf16* gelub   = (bf16*)alloc((size_t)ROWS * MLPD * 2);
  bf16* poolb   = (bf16*)alloc((size_t)BB * DIM * 2);
  (void)ws_size; (void)in_sizes; (void)n_in; (void)out_size;

  // weight conversion (each call; no cross-call state)
  {
    long n;
    n = (long)DEPTH * DIM * 3 * DIM;
    k_f32_to_bf16<<<(unsigned)cdiv(n, 256), 256, 0, stream>>>(w_qkv, wqkv_b, n);
    n = (long)DEPTH * DIM * DIM;
    k_f32_to_bf16<<<(unsigned)cdiv(n, 256), 256, 0, stream>>>(w_out, wout_b, n);
    n = (long)DEPTH * DIM * MLPD;
    k_f32_to_bf16<<<(unsigned)cdiv(n, 256), 256, 0, stream>>>(w1, w1_b, n);
    n = (long)DEPTH * MLPD * DIM;
    k_f32_to_bf16<<<(unsigned)cdiv(n, 256), 256, 0, stream>>>(w2, w2_b, n);
    n = (long)DIM * FEAT;
    k_f32_to_bf16<<<(unsigned)cdiv(n, 256), 256, 0, stream>>>(w_head, whead_b, n);
  }

  k_build_x<<<(unsigned)cdiv((long)ROWS * DIM, 256), 256, 0, stream>>>(x_in, cls, x0);

  const float attn_scale = 0.125f;  // 64^-0.5
  const unsigned gy = (unsigned)cdiv(ROWS, BM);        // 50
  const unsigned gyA = (unsigned)cdiv(NTOK, BM);       // 2

  for (int l = 0; l < DEPTH; ++l) {
    // LN1 -> bf16
    k_layernorm_bf16<<<ROWS, 256, 0, stream>>>(x0, DIM, ln1_g + l * DIM, ln1_b + l * DIM,
                                               hbuf, ROWS);
    // qkv = h @ w_qkv[l]   [6304 x 2304]
    gemm_bf16<0, false><<<dim3(3 * DIM / BN, gy, 1), 128, 0, stream>>>(
        hbuf, DIM, 0, wqkv_b + (long)l * DIM * 3 * DIM, 3 * DIM, 0,
        qkvbuf, nullptr, 3 * DIM, 0, nullptr, nullptr, ROWS, 3 * DIM, DIM, 1.0f);
    // split heads
    k_split_qkv<<<(unsigned)cdiv((long)ZBH * NPAD * DHEAD, 256), 256, 0, stream>>>(
        qkvbuf, qb, kb, vb);
    // scores = scale * q @ k^T  (batched, B col-major)
    gemm_bf16<0, true><<<dim3((unsigned)cdiv(NTOK, BN), gyA, ZBH), 128, 0, stream>>>(
        qb, DHEAD, (long)NTOK * DHEAD, kb, DHEAD, (long)NTOK * DHEAD,
        scores, nullptr, NPAD, (long)NTOK * NPAD, nullptr, nullptr,
        NTOK, NTOK, DHEAD, attn_scale);
    // softmax rows
    k_softmax<<<(unsigned)((long)ZBH * NTOK), 256, 0, stream>>>(scores);
    // re-attention mix + head-norm -> bf16 padded
    k_reattn<<<(unsigned)cdiv((long)BB * NTOK * NPAD, 256), 256, 0, stream>>>(
        scores, re_w + l * HEADS * HEADS, re_g + l * HEADS, re_b + l * HEADS, attn_n);
    // out = attn @ v  (batched)
    gemm_bf16<0, false><<<dim3(1, gyA, ZBH), 128, 0, stream>>>(
        attn_n, NPAD, (long)NTOK * NPAD, vb, DHEAD, (long)NPAD * DHEAD,
        aout, nullptr, DHEAD, (long)NTOK * DHEAD, nullptr, nullptr,
        NTOK, DHEAD, NPAD, 1.0f);
    // merge heads -> bf16 [6304 x 768]
    k_merge_heads<<<(unsigned)cdiv((long)ROWS * DIM, 256), 256, 0, stream>>>(aout, aout_b);
    // x1 = x0 + aout_b @ w_out[l] + b_out[l]
    gemm_bf16<3, false><<<dim3(DIM / BN, gy, 1), 128, 0, stream>>>(
        aout_b, DIM, 0, wout_b + (long)l * DIM * DIM, DIM, 0,
        x1, nullptr, DIM, 0, b_out + l * DIM, x0, ROWS, DIM, DIM, 1.0f);
    // LN2 -> bf16
    k_layernorm_bf16<<<ROWS, 256, 0, stream>>>(x1, DIM, ln2_g + l * DIM, ln2_b + l * DIM,
                                               hbuf, ROWS);
    // gelu(h @ w1 + b1) -> bf16
    gemm_bf16<2, false><<<dim3(MLPD / BN, gy, 1), 128, 0, stream>>>(
        hbuf, DIM, 0, w1_b + (long)l * DIM * MLPD, MLPD, 0,
        nullptr, gelub, MLPD, 0, b1 + l * MLPD, nullptr, ROWS, MLPD, DIM, 1.0f);
    // x0 = x1 + gelu @ w2 + b2
    gemm_bf16<3, false><<<dim3(DIM / BN, gy, 1), 128, 0, stream>>>(
        gelub, MLPD, 0, w2_b + (long)l * MLPD * DIM, DIM, 0,
        x0, nullptr, DIM, 0, b2 + l * DIM, x1, ROWS, DIM, MLPD, 1.0f);
  }

  // pooled = x0[:, 0]; final layernorm -> bf16
  k_layernorm_bf16<<<BB, 256, 0, stream>>>(x0, (long)NTOK * DIM, lnf_g, lnf_b, poolb, BB);
  // out = pooled_ln @ w_head + b_head   [32 x 1000]
  gemm_bf16<1, false><<<dim3((unsigned)cdiv(FEAT, BN), 1, 1), 128, 0, stream>>>(
      poolb, DIM, 0, whead_b, FEAT, 0,
      outp, nullptr, FEAT, 0, b_head, nullptr, BB, FEAT, DIM, 1.0f);
}
